// DeepSeekEncoderLayer_27384711480194
// MI455X (gfx1250) — matrix operand order repaired and checked by reading, not compiled, not run on hardware
//
#include <hip/hip_runtime.h>
#include <hip/hip_bf16.h>
#include <math.h>

// ---------------------------------------------------------------------------
// CDNA5 (gfx1250) wave32 WMMA transformer encoder layer.
// GEMMs: v_wmma_f32_16x16x32_bf16, LDS-staged 128x128 tiles,
// async global->LDS (GLOBAL_LOAD_ASYNC_TO_LDS_B128) for A tiles when available.
// ---------------------------------------------------------------------------

typedef __attribute__((ext_vector_type(16))) __bf16 v16bf;
typedef __attribute__((ext_vector_type(8)))  float  v8f;
typedef __attribute__((ext_vector_type(2)))  __bf16 bf16x2;
typedef __attribute__((ext_vector_type(4)))  __bf16 bf16x4;
typedef int v4i_t __attribute__((vector_size(16)));   // async-LDS builtin operand type

#define BB    4
#define SS    2048
#define EE    1024
#define HH    16
#define DD    64
#define FFN   4096
#define HALFW 128        // WINDOW/2

#define ASTR 40          // LDS row stride (elems): 80B rows -> 16B-aligned chunks
#define BSTR 40

#if defined(__gfx1250__) && __has_builtin(__builtin_amdgcn_global_load_async_to_lds_b128)
#define HAS_ASYNC_LDS 1
#else
#define HAS_ASYNC_LDS 0
#endif

static __device__ __forceinline__ v8f wmma_bf16(v16bf a, v16bf b, v8f c) {
  // D = A(16x32 bf16) * B(32x16 bf16) + C(16x16 f32)
  return __builtin_amdgcn_wmma_f32_16x16x32_bf16(false, a, false, b,
                                                 (short)0, c, false, false);
}

static __device__ __forceinline__ void async_g2l_16B(const __bf16* g, __bf16* l) {
#if HAS_ASYNC_LDS
  __builtin_amdgcn_global_load_async_to_lds_b128((v4i_t*)g, (v4i_t*)l, 0, 0);
#else
  (void)g; (void)l;
#endif
}

static __device__ __forceinline__ void wait_async_lds() {
#if HAS_ASYNC_LDS
#if __has_builtin(__builtin_amdgcn_s_wait_asynccnt)
  __builtin_amdgcn_s_wait_asynccnt(0);
#else
  asm volatile("s_wait_asynccnt 0x0" ::: "memory");
#endif
#endif
}

// ---------------------------------------------------------------------------
// f32 -> bf16 conversion (weights)
// ---------------------------------------------------------------------------
__global__ __launch_bounds__(256) void cvt_kernel(const float* __restrict__ in,
                                                  __bf16* __restrict__ out, int n) {
  int i = blockIdx.x * 256 + threadIdx.x;
  if (i < n) out[i] = (__bf16)in[i];
}

// ---------------------------------------------------------------------------
// LayerNorm (f32 in, bf16 out). One block of 256 threads per row of 1024.
// ---------------------------------------------------------------------------
__global__ __launch_bounds__(256) void ln_kernel(const float* __restrict__ x,
                                                 const float* __restrict__ gamma,
                                                 const float* __restrict__ beta,
                                                 __bf16* __restrict__ out) {
  const int row = blockIdx.x;
  const int tid = threadIdx.x;
  const float* xr = x + (size_t)row * EE;
  float v[4], s = 0.f, s2 = 0.f;
#pragma unroll
  for (int i = 0; i < 4; ++i) {
    v[i] = xr[tid + i * 256];
    s += v[i];
    s2 += v[i] * v[i];
  }
#pragma unroll
  for (int off = 16; off >= 1; off >>= 1) {
    s  += __shfl_xor(s,  off, 32);
    s2 += __shfl_xor(s2, off, 32);
  }
  __shared__ float sh1[8], sh2[8];
  if ((tid & 31) == 0) { sh1[tid >> 5] = s; sh2[tid >> 5] = s2; }
  __syncthreads();
  float ts = 0.f, t2 = 0.f;
#pragma unroll
  for (int w = 0; w < 8; ++w) { ts += sh1[w]; t2 += sh2[w]; }
  const float mu   = ts * (1.f / EE);
  const float var  = t2 * (1.f / EE) - mu * mu;
  const float rstd = rsqrtf(var + 1e-5f);
#pragma unroll
  for (int i = 0; i < 4; ++i) {
    int c = tid + i * 256;
    out[(size_t)row * EE + c] = (__bf16)((v[i] - mu) * rstd * gamma[c] + beta[c]);
  }
}

// ---------------------------------------------------------------------------
// Tile staging helpers (block = 256 threads)
// A tile: 128 rows x 32 cols bf16 into As[128][ASTR]
// B tile: 32 rows x 128 cols bf16, stored transposed into Bs[n][k]
// ---------------------------------------------------------------------------
static __device__ __forceinline__ void stage_A_tile(const __bf16* __restrict__ A,
                                                    __bf16* As, int m0, int k0,
                                                    int K, int tid) {
#if HAS_ASYNC_LDS
#pragma unroll
  for (int i = 0; i < 2; ++i) {          // 512 chunks of 16B, 2 per thread
    int c = tid + i * 256;
    int r = c >> 2, col = (c & 3) << 3;
    async_g2l_16B(A + (size_t)(m0 + r) * K + k0 + col, &As[r * ASTR + col]);
  }
#else
#pragma unroll
  for (int i = 0; i < 4; ++i) {
    int c = tid + i * 256;
    int r = c >> 3, col = (c & 7) << 2;
    bf16x4 t = *(const bf16x4*)(A + (size_t)(m0 + r) * K + k0 + col);
    *(bf16x4*)(&As[r * ASTR + col]) = t;
  }
#endif
}

static __device__ __forceinline__ void stage_B_tileT(const __bf16* __restrict__ Bw,
                                                     __bf16* Bs, int n0, int k0,
                                                     int N, int tid) {
#pragma unroll
  for (int i = 0; i < 4; ++i) {          // 1024 chunks of 4 elems, 4 per thread
    int c = tid + i * 256;
    int kr = c >> 5, nc = (c & 31) << 2;
    bf16x4 t = *(const bf16x4*)(Bw + (size_t)(k0 + kr) * N + n0 + nc);
    Bs[(nc + 0) * BSTR + kr] = t.x;
    Bs[(nc + 1) * BSTR + kr] = t.y;
    Bs[(nc + 2) * BSTR + kr] = t.z;
    Bs[(nc + 3) * BSTR + kr] = t.w;
  }
}

// A fragment: 16x32 bf16 (lanes 0-15: K 0..15; lanes 16-31: K 16..31)
static __device__ __forceinline__ v16bf load_afrag(const __bf16* As, int row, int hi) {
  v16bf af;
  int khalf = hi * 16;
#pragma unroll
  for (int jj = 0; jj < 8; ++jj) {
    int j = jj * 2;
    int k = khalf + j;
    bf16x2 t = *(const bf16x2*)(&As[row * ASTR + k]);
    af[j] = t.x; af[j + 1] = t.y;
  }
  return af;
}

// B fragment: 32x16 bf16 from transposed Bs[n][k] (lanes 0-15: K 0..15; 16-31: 16..31)
static __device__ __forceinline__ v16bf load_bfrag(const __bf16* Bs, int coln, int hi) {
  v16bf bf_;
  int kb2 = hi * 16;
#pragma unroll
  for (int g = 0; g < 4; ++g) {
    bf16x4 t = *(const bf16x4*)(&Bs[coln * BSTR + kb2 + g * 4]);
    bf_[g * 4 + 0] = t.x; bf_[g * 4 + 1] = t.y;
    bf_[g * 4 + 2] = t.z; bf_[g * 4 + 3] = t.w;
  }
  return bf_;
}

// ---------------------------------------------------------------------------
// WMMA GEMM: C[M,N] = A[M,K](bf16) @ Bw[K,N](bf16) + bias (+ residual)
// Block = 256 threads (8 waves). Tile 128x128, BK = 32. Wave grid 4x2;
// each wave computes a 32x64 region as 2x4 WMMA 16x16 tiles.
// ---------------------------------------------------------------------------
template <bool OUT_BF16, bool RESID>
__global__ __launch_bounds__(256) void gemm_kernel(
    const __bf16* __restrict__ A, const __bf16* __restrict__ Bw,
    const float* __restrict__ bias, const float* __restrict__ resid,
    void* __restrict__ Cout, int M, int N, int K) {
  __shared__ __bf16 As[128 * ASTR];
  __shared__ __bf16 Bs[128 * BSTR];   // transposed: Bs[n][k]
  const int tid  = threadIdx.x;
  const int lane = tid & 31, wid = tid >> 5;
  const int hi = lane >> 4, lo = lane & 15;
  const int m0 = blockIdx.y * 128, n0 = blockIdx.x * 128;
  const int wm = wid >> 1, wn = wid & 1;

  v8f acc[2][4];
#pragma unroll
  for (int mi = 0; mi < 2; ++mi)
#pragma unroll
    for (int ni = 0; ni < 4; ++ni)
#pragma unroll
      for (int r = 0; r < 8; ++r) acc[mi][ni][r] = 0.f;

  for (int k0 = 0; k0 < K; k0 += 32) {
    stage_A_tile(A, As, m0, k0, K, tid);
    stage_B_tileT(Bw, Bs, n0, k0, N, tid);
    if (k0 + 32 < K)   // global_prefetch_b8 for next A tile
      __builtin_prefetch(A + (size_t)(m0 + (tid >> 3)) * K + k0 + 32, 0, 0);
    wait_async_lds();
    __syncthreads();

    v16bf af[2], bf_[4];
#pragma unroll
    for (int mi = 0; mi < 2; ++mi)
      af[mi] = load_afrag(As, wm * 32 + mi * 16 + lo, hi);
#pragma unroll
    for (int ni = 0; ni < 4; ++ni)
      bf_[ni] = load_bfrag(Bs, wn * 64 + ni * 16 + lo, hi);
#pragma unroll
    for (int mi = 0; mi < 2; ++mi)
#pragma unroll
      for (int ni = 0; ni < 4; ++ni)
        acc[mi][ni] = wmma_bf16(af[mi], bf_[ni], acc[mi][ni]);
    __syncthreads();
  }

  // epilogue: bias (+residual). C layout: elem r -> row r+hi*8, col lo.
#pragma unroll
  for (int mi = 0; mi < 2; ++mi)
#pragma unroll
    for (int ni = 0; ni < 4; ++ni) {
      int col = n0 + wn * 64 + ni * 16 + lo;
      float bcol = bias[col];
#pragma unroll
      for (int r = 0; r < 8; ++r) {
        int row = m0 + wm * 32 + mi * 16 + r + hi * 8;
        float vv = acc[mi][ni][r] + bcol;
        if constexpr (RESID) vv += resid[(size_t)row * N + col];
        if constexpr (OUT_BF16)
          ((__bf16*)Cout)[(size_t)row * N + col] = (__bf16)vv;
        else
          ((float*)Cout)[(size_t)row * N + col] = vv;
      }
    }
}

// ---------------------------------------------------------------------------
// Fused GEGLU GEMM: Out = gelu(A@Bg + bg) * (A@Bv + bv), bf16 out.
// ---------------------------------------------------------------------------
__global__ __launch_bounds__(256) void glu_gemm_kernel(
    const __bf16* __restrict__ A, const __bf16* __restrict__ Bg,
    const __bf16* __restrict__ Bv, const float* __restrict__ bg,
    const float* __restrict__ bv, __bf16* __restrict__ Out,
    int M, int N, int K) {
  __shared__ __bf16 As[128 * ASTR];
  __shared__ __bf16 Bsg[128 * BSTR];
  __shared__ __bf16 Bsv[128 * BSTR];
  const int tid  = threadIdx.x;
  const int lane = tid & 31, wid = tid >> 5;
  const int hi = lane >> 4, lo = lane & 15;
  const int m0 = blockIdx.y * 128, n0 = blockIdx.x * 128;
  const int wm = wid >> 1, wn = wid & 1;

  v8f accg[2][4], accv[2][4];
#pragma unroll
  for (int mi = 0; mi < 2; ++mi)
#pragma unroll
    for (int ni = 0; ni < 4; ++ni)
#pragma unroll
      for (int r = 0; r < 8; ++r) { accg[mi][ni][r] = 0.f; accv[mi][ni][r] = 0.f; }

  for (int k0 = 0; k0 < K; k0 += 32) {
    stage_A_tile(A, As, m0, k0, K, tid);
    stage_B_tileT(Bg, Bsg, n0, k0, N, tid);
    stage_B_tileT(Bv, Bsv, n0, k0, N, tid);
    wait_async_lds();
    __syncthreads();

    v16bf af[2], bfg[4], bfv[4];
#pragma unroll
    for (int mi = 0; mi < 2; ++mi)
      af[mi] = load_afrag(As, wm * 32 + mi * 16 + lo, hi);
#pragma unroll
    for (int ni = 0; ni < 4; ++ni) {
      bfg[ni] = load_bfrag(Bsg, wn * 64 + ni * 16 + lo, hi);
      bfv[ni] = load_bfrag(Bsv, wn * 64 + ni * 16 + lo, hi);
    }
#pragma unroll
    for (int mi = 0; mi < 2; ++mi)
#pragma unroll
      for (int ni = 0; ni < 4; ++ni) {
        accg[mi][ni] = wmma_bf16(af[mi], bfg[ni], accg[mi][ni]);
        accv[mi][ni] = wmma_bf16(af[mi], bfv[ni], accv[mi][ni]);
      }
    __syncthreads();
  }

#pragma unroll
  for (int mi = 0; mi < 2; ++mi)
#pragma unroll
    for (int ni = 0; ni < 4; ++ni) {
      int col = n0 + wn * 64 + ni * 16 + lo;
      float bgc = bg[col], bvc = bv[col];
#pragma unroll
      for (int r = 0; r < 8; ++r) {
        int row = m0 + wm * 32 + mi * 16 + r + hi * 8;
        float g  = accg[mi][ni][r] + bgc;
        float vv = accv[mi][ni][r] + bvc;
        float ge = 0.5f * g * (1.f + erff(g * 0.70710678f));  // exact gelu
        Out[(size_t)row * N + col] = (__bf16)(ge * vv);
      }
    }
}

// ---------------------------------------------------------------------------
// RoPE + split: qkv bf16 [B*S, 3E] -> q,k,v bf16 [B,H,S,D]
// ---------------------------------------------------------------------------
__global__ __launch_bounds__(256) void rope_kernel(const __bf16* __restrict__ qkv,
                                                   __bf16* __restrict__ q,
                                                   __bf16* __restrict__ k,
                                                   __bf16* __restrict__ v) {
  int idx = blockIdx.x * 256 + threadIdx.x;   // over B*S*H*D
  int d = idx & 63;
  int h = (idx >> 6) & 15;
  int s = (idx >> 10) & 2047;
  int b = idx >> 21;
  size_t row = (size_t)(b * SS + s) * (3 * EE);
  int c0 = h * DD + d;
  float qv = (float)qkv[row + c0];
  float kv = (float)qkv[row + EE + c0];
  float vv = (float)qkv[row + 2 * EE + c0];
  int j = d & 31;                                   // inv_freq index
  float inv = __expf(-0.28782313662f * (float)j);   // 10000^(-2j/64)
  float ang = (float)s * inv;
  float sn, cs;
  __sincosf(ang, &sn, &cs);
  int dp = (d < 32) ? (d + 32) : (d - 32);
  float sgn = (d < 32) ? -1.f : 1.f;                // rotate_half sign
  float qp = (float)qkv[row + h * DD + dp];
  float kp = (float)qkv[row + EE + h * DD + dp];
  float qo = qv * cs + sgn * qp * sn;
  float ko = kv * cs + sgn * kp * sn;
  size_t o = ((size_t)(b * HH + h) * SS + s) * DD + d;
  q[o] = (__bf16)qo;
  k[o] = (__bf16)ko;
  v[o] = (__bf16)vv;
}

// ---------------------------------------------------------------------------
// Sliding-window flash attention, one wave per 16-row query tile.
// 8 waves/block -> 128 query rows per block, one (b,h) per block group.
// 9 key-blocks of 32 cover the aligned |i-j|<=128 window span.
// Q@K^T and P@V via bf16 WMMA; P relaid out via per-wave LDS.
// ---------------------------------------------------------------------------
__global__ __launch_bounds__(256) void attn_kernel(const __bf16* __restrict__ Q,
                                                   const __bf16* __restrict__ Kk,
                                                   const __bf16* __restrict__ V,
                                                   __bf16* __restrict__ Out) {
  __shared__ __bf16 Pl[8][16 * 34];                  // per-wave P staging
  const int tid  = threadIdx.x;
  const int lane = tid & 31, wid = tid >> 5;
  const int hi = lane >> 4, lo = lane & 15;
  const int bh   = blockIdx.x >> 4;                  // S/128 = 16 tiles per (b,h)
  const int tile = blockIdx.x & 15;
  const int q0   = tile * 128 + wid * 16;
  const size_t base = (size_t)bh * SS * DD;

  // load Q fragments (two K=32 chunks across D=64)
  v16bf aq[2];
#pragma unroll
  for (int kc = 0; kc < 2; ++kc) {
    int row = q0 + lo;
#pragma unroll
    for (int jj = 0; jj < 8; ++jj) {
      int j = jj * 2;
      int d = kc * 32 + hi * 16 + j;
      bf16x2 t = *(const bf16x2*)(Q + base + (size_t)row * DD + d);
      aq[kc][j] = t.x; aq[kc][j + 1] = t.y;
    }
  }

  float mrun[8], lrun[8];
  v8f o[4];
#pragma unroll
  for (int r = 0; r < 8; ++r) { mrun[r] = -1e30f; lrun[r] = 0.f; }
#pragma unroll
  for (int t4 = 0; t4 < 4; ++t4)
#pragma unroll
    for (int r = 0; r < 8; ++r) o[t4][r] = 0.f;

  const int kstart = (q0 - HALFW) & ~31;             // floor-align to 32
  for (int st = 0; st < 9; ++st) {
    const int kb = kstart + st * 32;
    // K^T fragments: B role 32(d) x 16(keys), two key tiles x two d chunks
    v16bf bk[2][2];
#pragma unroll
    for (int ni = 0; ni < 2; ++ni) {
      int key = kb + ni * 16 + lo;
      int keyc = min(max(key, 0), SS - 1);
#pragma unroll
      for (int kc = 0; kc < 2; ++kc)
#pragma unroll
        for (int g = 0; g < 4; ++g) {
          int d = kc * 32 + hi * 16 + g * 4;
          bf16x4 t = *(const bf16x4*)(Kk + base + (size_t)keyc * DD + d);
          bk[ni][kc][g * 4 + 0] = t.x; bk[ni][kc][g * 4 + 1] = t.y;
          bk[ni][kc][g * 4 + 2] = t.z; bk[ni][kc][g * 4 + 3] = t.w;
        }
    }
    v8f sc[2];
#pragma unroll
    for (int ni = 0; ni < 2; ++ni) {
#pragma unroll
      for (int r = 0; r < 8; ++r) sc[ni][r] = 0.f;
      sc[ni] = wmma_bf16(aq[0], bk[ni][0], sc[ni]);
      sc[ni] = wmma_bf16(aq[1], bk[ni][1], sc[ni]);
    }
    // scale 1/sqrt(64) + window mask
#pragma unroll
    for (int ni = 0; ni < 2; ++ni) {
      int key = kb + ni * 16 + lo;
#pragma unroll
      for (int r = 0; r < 8; ++r) {
        int qr = q0 + r + hi * 8;
        int dd = qr - key; dd = dd < 0 ? -dd : dd;
        bool ok = (key >= 0) && (key < SS) && (dd <= HALFW);
        sc[ni][r] = ok ? sc[ni][r] * 0.125f : -1e9f;
      }
    }
    // online softmax update (row reductions across the 16-lane column group)
#pragma unroll
    for (int r = 0; r < 8; ++r) {
      float t = fmaxf(sc[0][r], sc[1][r]);
#pragma unroll
      for (int off = 1; off < 16; off <<= 1) t = fmaxf(t, __shfl_xor(t, off, 32));
      float mn = fmaxf(mrun[r], t);
      float alpha = __expf(mrun[r] - mn);
      mrun[r] = mn;
      float ps = 0.f;
#pragma unroll
      for (int ni = 0; ni < 2; ++ni) {
        float p = __expf(sc[ni][r] - mn);
        sc[ni][r] = p;
        ps += p;
      }
#pragma unroll
      for (int off = 1; off < 16; off <<= 1) ps += __shfl_xor(ps, off, 32);
      lrun[r] = lrun[r] * alpha + ps;
#pragma unroll
      for (int t4 = 0; t4 < 4; ++t4) o[t4][r] *= alpha;
    }
    // stage P (C layout -> LDS row-major [16][32], stride 34)
    __bf16* pl = &Pl[wid][0];
#pragma unroll
    for (int ni = 0; ni < 2; ++ni)
#pragma unroll
      for (int r = 0; r < 8; ++r)
        pl[(r + hi * 8) * 34 + ni * 16 + lo] = (__bf16)sc[ni][r];
    __syncthreads();
    // reload P as A fragment (16 rows x 32 keys)
    v16bf ap;
#pragma unroll
    for (int jj = 0; jj < 8; ++jj) {
      int j = jj * 2;
      int kk = hi * 16 + j;
      bf16x2 t = *(const bf16x2*)(&pl[lo * 34 + kk]);
      ap[j] = t.x; ap[j + 1] = t.y;
    }
    // V fragments (B role 32 keys x 16 d) and accumulate O
#pragma unroll
    for (int t4 = 0; t4 < 4; ++t4) {
      v16bf bv;
      int dcol = t4 * 16 + lo;
#pragma unroll
      for (int j = 0; j < 16; ++j) {
        int key = kb + hi * 16 + j;
        int keyc = min(max(key, 0), SS - 1);
        bv[j] = V[base + (size_t)keyc * DD + dcol];
      }
      o[t4] = wmma_bf16(ap, bv, o[t4]);
    }
    __syncthreads();
  }

  // normalize and write [B,S,E] bf16 (col = h*64 + d)
  const int b = bh >> 4, h = bh & 15;
#pragma unroll
  for (int t4 = 0; t4 < 4; ++t4) {
    int col = h * 64 + t4 * 16 + lo;
#pragma unroll
    for (int r = 0; r < 8; ++r) {
      int qr = q0 + r + hi * 8;
      Out[((size_t)(b * SS + qr)) * EE + col] = (__bf16)(o[t4][r] / lrun[r]);
    }
  }
}

// ---------------------------------------------------------------------------
// host launch
// ---------------------------------------------------------------------------
extern "C" void kernel_launch(void* const* d_in, const int* in_sizes, int n_in,
                              void* d_out, int out_size, void* d_ws, size_t ws_size,
                              hipStream_t stream) {
  (void)in_sizes; (void)n_in; (void)out_size; (void)ws_size;
  const float* src    = (const float*)d_in[0];
  const float* Wqkv   = (const float*)d_in[1];
  const float* bqkv   = (const float*)d_in[2];
  const float* Wout   = (const float*)d_in[3];
  const float* bout   = (const float*)d_in[4];
  const float* gamma1 = (const float*)d_in[5];
  const float* beta1  = (const float*)d_in[6];
  const float* gamma2 = (const float*)d_in[7];
  const float* beta2  = (const float*)d_in[8];
  const float* Wg     = (const float*)d_in[9];
  const float* bg     = (const float*)d_in[10];
  const float* Wv     = (const float*)d_in[11];
  const float* bv     = (const float*)d_in[12];
  const float* Wo     = (const float*)d_in[13];
  const float* bo     = (const float*)d_in[14];

  char* ws = (char*)d_ws;
  // bf16 weights: 16,777,216 elems = 32 MB at offset 0
  __bf16* wqkv_b = (__bf16*)ws;
  __bf16* wout_b = wqkv_b + 3145728;   // E*3E
  __bf16* wg_b   = wqkv_b + 4194304;
  __bf16* wv_b   = wqkv_b + 8388608;
  __bf16* wo_b   = wqkv_b + 12582912;
  // region1 [32MB,48MB): h1 bf16, later attn bf16 (h1 dead)
  __bf16* h1   = (__bf16*)(ws + (size_t)32 * 1024 * 1024);
  __bf16* attn = h1;
  // region2 [48MB,96MB): qkv bf16 (48MB); later x1 f32 (32MB) + h2 bf16 (16MB)
  __bf16* qkvb = (__bf16*)(ws + (size_t)48 * 1024 * 1024);
  float*  x1   = (float*)(ws + (size_t)48 * 1024 * 1024);
  __bf16* h2   = (__bf16*)(ws + (size_t)80 * 1024 * 1024);
  // region3 [96MB,160MB): q/k/v bf16 (16MB each); later p bf16 (64MB)
  __bf16* qb = (__bf16*)(ws + (size_t)96 * 1024 * 1024);
  __bf16* kb = (__bf16*)(ws + (size_t)112 * 1024 * 1024);
  __bf16* vb = (__bf16*)(ws + (size_t)128 * 1024 * 1024);
  __bf16* pb = (__bf16*)(ws + (size_t)96 * 1024 * 1024);

  const int M = BB * SS;  // 8192

  // 1) weights -> bf16
  int n;
  n = EE * 3 * EE; cvt_kernel<<<(n + 255) / 256, 256, 0, stream>>>(Wqkv, wqkv_b, n);
  n = EE * EE;     cvt_kernel<<<(n + 255) / 256, 256, 0, stream>>>(Wout, wout_b, n);
  n = EE * FFN;    cvt_kernel<<<(n + 255) / 256, 256, 0, stream>>>(Wg,   wg_b,   n);
  n = EE * FFN;    cvt_kernel<<<(n + 255) / 256, 256, 0, stream>>>(Wv,   wv_b,   n);
  n = FFN * EE;    cvt_kernel<<<(n + 255) / 256, 256, 0, stream>>>(Wo,   wo_b,   n);

  // 2) LN1
  ln_kernel<<<M, 256, 0, stream>>>(src, gamma1, beta1, h1);

  // 3) QKV projection: [8192,1024] @ [1024,3072] -> bf16
  gemm_kernel<true, false><<<dim3(3 * EE / 128, M / 128), 256, 0, stream>>>(
      h1, wqkv_b, bqkv, nullptr, qkvb, M, 3 * EE, EE);

  // 4) RoPE + split
  rope_kernel<<<(BB * SS * HH * DD) / 256, 256, 0, stream>>>(qkvb, qb, kb, vb);

  // 5) windowed attention
  attn_kernel<<<BB * HH * (SS / 128), 256, 0, stream>>>(qb, kb, vb, attn);

  // 6) out projection + residual -> x1 f32
  gemm_kernel<false, true><<<dim3(EE / 128, M / 128), 256, 0, stream>>>(
      attn, wout_b, bout, src, x1, M, EE, EE);

  // 7) LN2
  ln_kernel<<<M, 256, 0, stream>>>(x1, gamma2, beta2, h2);

  // 8) fused GEGLU: gelu(h2@Wg+bg) * (h2@Wv+bv) -> bf16 p [8192,4096]
  glu_gemm_kernel<<<dim3(FFN / 128, M / 128), 256, 0, stream>>>(
      h2, wg_b, wv_b, bg, bv, pb, M, FFN, EE);

  // 9) down projection + residual -> d_out f32
  gemm_kernel<false, true><<<dim3(EE / 128, M / 128), 256, 0, stream>>>(
      pb, wo_b, bo, x1, (float*)d_out, M, EE, FFN);
}
